// ExpandProject_58093727646123
// MI455X (gfx1250) — matrix operand order repaired and checked by reading, not compiled, not run on hardware
//
#include <hip/hip_runtime.h>
#include <cmath>

typedef __bf16 bf16_t;
typedef __attribute__((ext_vector_type(16))) __bf16 v16bf;
typedef __attribute__((ext_vector_type(8)))  __bf16 v8bf;
typedef __attribute__((ext_vector_type(4)))  __bf16 v4bf;
typedef __attribute__((ext_vector_type(8)))  float  v8f;
typedef __attribute__((ext_vector_type(4)))  float  v4f;
typedef __attribute__((ext_vector_type(4)))  int    v4i;

#define HDIM 1024      // K
#define IDIM 4096      // N
#define MTOT 16384     // B*S
#define TM 128
#define TN 128
#define TK 32

#if defined(__gfx1250__) && __has_builtin(__builtin_amdgcn_global_load_async_to_lds_b128)
#define USE_ASYNC_LDS 1
#else
#define USE_ASYNC_LDS 0
#endif

#if USE_ASYNC_LDS
// Per-lane async 16B copy global -> LDS (tracked by ASYNCcnt, no VGPR data).
// Global generic pointer == AS(1) value; generic LDS pointer's low 32 bits
// are the LDS offset (ISA aperture rule), so integer casts are exact.
// Builtin signature (per hipcc diagnostic): (int4 AS1*, int4 AS3*, Imm, Imm).
template <int OFF>
__device__ __forceinline__ void async_b128(const void* g, void* l) {
  __builtin_amdgcn_global_load_async_to_lds_b128(
      (__attribute__((address_space(1))) v4i*)(uintptr_t)g,
      (__attribute__((address_space(3))) v4i*)(unsigned int)(uintptr_t)l,
      OFF, 0);
}
__device__ __forceinline__ void wait_async0() {
#if __has_builtin(__builtin_amdgcn_s_wait_asynccnt)
  __builtin_amdgcn_s_wait_asynccnt(0);
#else
  asm volatile("s_wait_asynccnt 0" ::: "memory");
#endif
}
#endif

// ---------------------------------------------------------------------------
// Kernel 1: row-wise LayerNorm, output bf16 activations (one block per row)
// ---------------------------------------------------------------------------
__global__ __launch_bounds__(256) void ln_to_bf16_kernel(
    const float* __restrict__ x, const float* __restrict__ gamma,
    const float* __restrict__ beta, bf16_t* __restrict__ xn) {
  const int row = blockIdx.x;
  const int tid = threadIdx.x;
  const float* xr = x + (size_t)row * HDIM;

  v4f v = *(const v4f*)(xr + tid * 4);
  float s  = v.x + v.y + v.z + v.w;
  float sq = v.x * v.x + v.y * v.y + v.z * v.z + v.w * v.w;

  // wave32 reduction
  #pragma unroll
  for (int off = 16; off > 0; off >>= 1) {
    s  += __shfl_xor(s,  off, 32);
    sq += __shfl_xor(sq, off, 32);
  }

  __shared__ float ssum[8];
  __shared__ float ssq[8];
  const int wave = tid >> 5;
  const int lane = tid & 31;
  if (lane == 0) { ssum[wave] = s; ssq[wave] = sq; }
  __syncthreads();
  if (tid == 0) {
    float S = 0.f, Q = 0.f;
    #pragma unroll
    for (int i = 0; i < 8; ++i) { S += ssum[i]; Q += ssq[i]; }
    ssum[0] = S; ssq[0] = Q;
  }
  __syncthreads();

  const float mean = ssum[0] * (1.0f / HDIM);
  const float var  = ssq[0] * (1.0f / HDIM) - mean * mean;
  const float rstd = rsqrtf(var + 1e-7f);

  v4bf o;
  #pragma unroll
  for (int j = 0; j < 4; ++j) {
    const int c = tid * 4 + j;
    float xv = (j == 0 ? v.x : j == 1 ? v.y : j == 2 ? v.z : v.w);
    float yn = (xv - mean) * rstd * gamma[c] + beta[c];
    o[j] = (bf16_t)yn;
  }
  *(v4bf*)(xn + (size_t)row * HDIM + tid * 4) = o;
}

// ---------------------------------------------------------------------------
// Kernel 2: W[H][I] (f32) -> Wt[I][H] (bf16), LDS-tiled transpose
// ---------------------------------------------------------------------------
__global__ __launch_bounds__(256) void w_transpose_bf16_kernel(
    const float* __restrict__ W, bf16_t* __restrict__ wt) {
  __shared__ float tile[32][33];
  const int n0 = blockIdx.x * 32;  // along I
  const int k0 = blockIdx.y * 32;  // along H
  const int tx = threadIdx.x & 31;
  const int ty = threadIdx.x >> 5; // 0..7
  #pragma unroll
  for (int r = ty; r < 32; r += 8)
    tile[r][tx] = W[(size_t)(k0 + r) * IDIM + n0 + tx];
  __syncthreads();
  #pragma unroll
  for (int r = ty; r < 32; r += 8)
    wt[(size_t)(n0 + r) * HDIM + k0 + tx] = (bf16_t)tile[tx][r];
}

// ---------------------------------------------------------------------------
// Kernel 3: GEMM (bf16 WMMA, f32 acc) + bias + erf-GELU
//   out[m][n] = gelu( sum_k xn[m][k] * wt[n][k] + bias[n] )
//   WG: 256 threads = 8 waves (2M x 4N), WG tile 128x128, wave tile 64x32.
//   Double-buffered LDS staging via async global->LDS (CDNA5 path).
// ---------------------------------------------------------------------------
__global__ __launch_bounds__(256) void gemm_bias_gelu_kernel(
    const bf16_t* __restrict__ xn, const bf16_t* __restrict__ wt,
    const float* __restrict__ bias, float* __restrict__ out) {
  __shared__ __attribute__((aligned(64))) bf16_t As[2][TM][TK];
  __shared__ __attribute__((aligned(64))) bf16_t Bs[2][TN][TK];

  const int tid   = threadIdx.x;
  const int wave  = tid >> 5;
  const int lane  = tid & 31;
  const int lrow  = lane & 15;   // N (for B/C) or M (for A) within 16
  const int half  = lane >> 4;   // lane half selects K/M sub-blocks
  const int waveM = wave & 1;    // 0..1  -> 64-row strip
  const int waveN = wave >> 1;   // 0..3  -> 32-col strip
  const int m0 = blockIdx.y * TM;
  const int n0 = blockIdx.x * TN;

  // cooperative tile copy: each thread moves 32B of A and 32B of B per step
  const int crow   = tid >> 1;         // 0..127
  const int cchunk = (tid & 1) * 16;   // 0 or 16 (bf16 elements)

  v8f acc[4][2];
  #pragma unroll
  for (int mi = 0; mi < 4; ++mi)
    #pragma unroll
    for (int ni = 0; ni < 2; ++ni)
      acc[mi][ni] = (v8f){0.f, 0.f, 0.f, 0.f, 0.f, 0.f, 0.f, 0.f};

  auto load_tiles = [&](int buf, int k0) {
    const bf16_t* ga = xn + (size_t)(m0 + crow) * HDIM + k0 + cchunk;
    const bf16_t* gb = wt + (size_t)(n0 + crow) * HDIM + k0 + cchunk;
    bf16_t* la = &As[buf][crow][cchunk];
    bf16_t* lb = &Bs[buf][crow][cchunk];
#if USE_ASYNC_LDS
    async_b128<0>(ga, la);
    async_b128<16>(ga, la);
    async_b128<0>(gb, lb);
    async_b128<16>(gb, lb);
#else
    *(v16bf*)la = *(const v16bf*)ga;
    *(v16bf*)lb = *(const v16bf*)gb;
#endif
  };

  constexpr int NK = HDIM / TK;  // 32 K-steps
  load_tiles(0, 0);
#if USE_ASYNC_LDS
  wait_async0();
#endif
  __syncthreads();

  int buf = 0;
  for (int kk = 0; kk < NK; ++kk) {
    if (kk + 1 < NK) load_tiles(buf ^ 1, (kk + 1) * TK);

    // A fragments: lane holds row M=lrow; v0..7 = K[half*16..+16) contiguous
    v16bf afrag[4];
    #pragma unroll
    for (int mi = 0; mi < 4; ++mi) {
      const bf16_t* ap = &As[buf][waveM * 64 + mi * 16 + lrow][0];
      union { v16bf v; v8bf h[2]; } ua;
      ua.h[0] = *(const v8bf*)(ap + half * 16);
      ua.h[1] = *(const v8bf*)(ap + half * 16 + 8);
      afrag[mi] = ua.v;
    }

    // B fragments: lane holds col N=lrow; v0..7 = K[half*16..+16) contiguous
    #pragma unroll
    for (int ni = 0; ni < 2; ++ni) {
      const bf16_t* bp = &Bs[buf][waveN * 32 + ni * 16 + lrow][0];
      v16bf bfrag = *(const v16bf*)(bp + half * 16);
      #pragma unroll
      for (int mi = 0; mi < 4; ++mi) {
        acc[mi][ni] = __builtin_amdgcn_wmma_f32_16x16x32_bf16(
            /*neg_a=*/false, afrag[mi], /*neg_b=*/false, bfrag,
            /*c_mod=*/(short)0, acc[mi][ni],
            /*reuse_a=*/false, /*reuse_b=*/false);
      }
    }
#if USE_ASYNC_LDS
    wait_async0();
#endif
    __syncthreads();
    buf ^= 1;
  }

  // Epilogue: C layout element r <-> M = half*8 + r, N = lrow
  const float inv_sqrt2 = 0.70710678118654752f;
  #pragma unroll
  for (int ni = 0; ni < 2; ++ni) {
    const int n = n0 + waveN * 32 + ni * 16 + lrow;
    const float bn = bias[n];
    #pragma unroll
    for (int mi = 0; mi < 4; ++mi) {
      #pragma unroll
      for (int r = 0; r < 8; ++r) {
        const int m = m0 + waveM * 64 + mi * 16 + half * 8 + r;
        float y = acc[mi][ni][r] + bn;
        out[(size_t)m * IDIM + n] = 0.5f * y * (1.0f + erff(y * inv_sqrt2));
      }
    }
  }
}

// ---------------------------------------------------------------------------
extern "C" void kernel_launch(void* const* d_in, const int* in_sizes, int n_in,
                              void* d_out, int out_size, void* d_ws, size_t ws_size,
                              hipStream_t stream) {
  const float* hs    = (const float*)d_in[0];  // [4,4096,1024]
  const float* gamma = (const float*)d_in[1];  // [1024]
  const float* beta  = (const float*)d_in[2];  // [1024]
  const float* W     = (const float*)d_in[3];  // [1024,4096]
  const float* bias  = (const float*)d_in[4];  // [4096]
  float* out = (float*)d_out;                  // [4,4096,4096]

  bf16_t* xn = (bf16_t*)d_ws;                                  // 32 MB
  bf16_t* wt = (bf16_t*)((char*)d_ws +
                         (size_t)MTOT * HDIM * sizeof(bf16_t)); // +8 MB

  ln_to_bf16_kernel<<<MTOT, 256, 0, stream>>>(hs, gamma, beta, xn);

  dim3 tg(IDIM / 32, HDIM / 32);
  w_transpose_bf16_kernel<<<tg, 256, 0, stream>>>(W, wt);

  dim3 gg(IDIM / TN, MTOT / TM);
  gemm_bias_gelu_kernel<<<gg, 256, 0, stream>>>(xn, wt, bias, out);
}
